// DIEN_3135326126573
// MI455X (gfx1250) — matrix operand order repaired and checked by reading, not compiled, not run on hardware
//
#include <hip/hip_runtime.h>
#include <hip/hip_bf16.h>
#include <cstdint>
#include <cstddef>

#define DEV __device__ __forceinline__

constexpr int Bsz = 4096;   // batch
constexpr int Tn  = 200;    // sequence length
constexpr int En  = 128;    // embedding dim
constexpr int Hn  = 128;    // hidden dim
constexpr int Gn  = 384;    // 3*H (gates)

typedef __attribute__((ext_vector_type(16))) __bf16 v16bf;
typedef __attribute__((ext_vector_type(8)))  float  v8f;

// ---------- scalar helpers ----------
DEV uint16_t f2bf(float f) {
  uint32_t u = __float_as_uint(f);
  uint32_t r = u + 0x7FFFu + ((u >> 16) & 1u);   // round-to-nearest-even
  return (uint16_t)(r >> 16);
}
DEV uint32_t pk2(float a, float b) {
  return (uint32_t)f2bf(a) | ((uint32_t)f2bf(b) << 16);
}
DEV float sigmoidf_(float x) { return 1.0f / (1.0f + expf(-x)); }

// ---------- WMMA fragment loaders (CDNA5 16x16x32 bf16 layouts, ISA 7.12.2) ----------
// A matrix 16x32 (MxK), row-major tile with row stride 128 elements.
// lane<16: row=lane, K = kbase+{0..7} (v0..3) and kbase+{16..23} (v4..7)
// lane>=16: row=lane-16, K = kbase+{8..15} and kbase+{24..31}
DEV v16bf load_a_frag(const uint16_t* tile, int kbase) {
  const int lane = threadIdx.x & 31;
  const int r = lane & 15, hi = lane >> 4;
  const uint16_t* p = tile + r * En + kbase + hi * 8;
  union { v16bf v; uint4 q[2]; } u;
  u.q[0] = *reinterpret_cast<const uint4*>(p);
  u.q[1] = *reinterpret_cast<const uint4*>(p + 16);
  return u.v;
}
// B matrix 32x16 (KxN) taken from row-major W[N][K] (K stride = 128 elements).
// lane<16: n=lane, K = kbase+{0..7} and kbase+{16..23}; lane>=16: n=lane-16, K = kbase+{8..15} and kbase+{24..31}
DEV v16bf load_b_frag(const uint16_t* W, int n0, int kbase) {
  const int lane = threadIdx.x & 31;
  const int n = n0 + (lane & 15), hi = lane >> 4;
  const uint16_t* p = W + n * 128 + kbase + hi * 8;
  union { v16bf v; uint4 q[2]; } u;
  u.q[0] = *reinterpret_cast<const uint4*>(p);
  u.q[1] = *reinterpret_cast<const uint4*>(p + 16);
  return u.v;
}

// 8 waves; 24 N-tiles of gi (Wih @ x) + 24 N-tiles of gh (Whh @ h); 6 tiles/wave;
// K = 128 -> 4 WMMAs per tile. Accumulators seeded with bias. Results to LDS.
DEV void gemm_tiles(const uint16_t* xs, const uint16_t* hs,
                    const uint16_t* Wih, const uint16_t* Whh,
                    const float* bih, const float* bhh,
                    float* gi_s, float* gh_s) {
  const int w    = threadIdx.x >> 5;
  const int lane = threadIdx.x & 31;
  const int hi   = lane >> 4;
  for (int i = 0; i < 6; ++i) {
    const int tt = w * 6 + i;            // uniform per wave -> EXEC stays all-ones
    const bool is_gi = tt < 24;
    const int n0 = (is_gi ? tt : tt - 24) * 16;
    const uint16_t* W   = is_gi ? Wih : Whh;
    const float* bias   = is_gi ? bih : bhh;
    const uint16_t* A   = is_gi ? xs : hs;
    const float bn = bias[n0 + (lane & 15)];
    v8f acc = {bn, bn, bn, bn, bn, bn, bn, bn};
#pragma unroll
    for (int kk = 0; kk < 4; ++kk) {
      v16bf af = load_a_frag(A, kk * 32);
      v16bf bf = load_b_frag(W, n0, kk * 32);
      acc = __builtin_amdgcn_wmma_f32_16x16x32_bf16(
          /*neg_a=*/false, af, /*neg_b=*/false, bf,
          /*c_mod=*/(short)0, acc, /*reuse_a=*/false, /*reuse_b=*/false);
    }
    float* out = is_gi ? gi_s : gh_s;
    const int n = n0 + (lane & 15);
#pragma unroll
    for (int r = 0; r < 8; ++r) out[(r + 8 * hi) * Gn + n] = acc[r];
  }
}

// ---------- scan 1: interest-extractor GRU step (fused gather + dual GEMM + gates + score) ----------
__global__ __launch_bounds__(256) void dien_gru1_step(
    const int* __restrict__ user_hist, const float* __restrict__ emb,
    const uint16_t* __restrict__ Wih, const uint16_t* __restrict__ Whh,
    const float* __restrict__ bih, const float* __restrict__ bhh,
    const float* __restrict__ wa, const float* __restrict__ ba,
    const float* __restrict__ h_in, float* __restrict__ h_out,
    float* __restrict__ scores, int t) {
  __shared__ alignas(16) uint16_t xs[16 * En];
  __shared__ alignas(16) uint16_t hs[16 * En];
  __shared__ alignas(16) float gi_s[16 * Gn];
  __shared__ alignas(16) float gh_s[16 * Gn];
  __shared__ float red[256];
  const int tid = threadIdx.x;
  const int b0  = blockIdx.x * 16;
  const int m   = tid >> 4;
  const int c   = (tid & 15) * 8;
  const int b   = b0 + m;

  // weights are tiny and shared by all blocks: keep them hot (global_prefetch_b8)
  __builtin_prefetch(Wih + tid * 192, 0, 3);
  __builtin_prefetch(Whh + tid * 192, 0, 3);

  {  // stage x_t (gathered embedding row) and h as bf16 into LDS
    const int idx = user_hist[b * Tn + t];
    const float4* xr = reinterpret_cast<const float4*>(emb + (size_t)idx * En + c);
    float4 x0 = xr[0], x1 = xr[1];
    uint32_t* xd = reinterpret_cast<uint32_t*>(&xs[m * En + c]);
    xd[0] = pk2(x0.x, x0.y); xd[1] = pk2(x0.z, x0.w);
    xd[2] = pk2(x1.x, x1.y); xd[3] = pk2(x1.z, x1.w);
    const float4* hr = reinterpret_cast<const float4*>(h_in + b * Hn + c);
    float4 h0 = hr[0], h1 = hr[1];
    uint32_t* hd = reinterpret_cast<uint32_t*>(&hs[m * En + c]);
    hd[0] = pk2(h0.x, h0.y); hd[1] = pk2(h0.z, h0.w);
    hd[2] = pk2(h1.x, h1.y); hd[3] = pk2(h1.z, h1.w);
  }
  __syncthreads();
  gemm_tiles(xs, hs, Wih, Whh, bih, bhh, gi_s, gh_s);
  __syncthreads();

  float sc = 0.f;
#pragma unroll
  for (int jj = 0; jj < 8; ++jj) {
    const int j = c + jj;
    const float r  = sigmoidf_(gi_s[m * Gn + j]       + gh_s[m * Gn + j]);
    const float z  = sigmoidf_(gi_s[m * Gn + 128 + j] + gh_s[m * Gn + 128 + j]);
    const float n  = tanhf(gi_s[m * Gn + 256 + j] + r * gh_s[m * Gn + 256 + j]);
    const float ho = h_in[b * Hn + j];
    const float hn = (1.f - z) * n + z * ho;
    h_out[b * Hn + j] = hn;
    sc += hn * wa[j];
  }
  red[tid] = sc;
  __syncthreads();
  if ((tid & 15) == 0) {                    // one lane per batch row: attention score
    float s = 0.f;
#pragma unroll
    for (int i = 0; i < 16; ++i) s += red[tid + i];
    scores[b * Tn + t] = s + ba[0];
  }
}

// ---------- attention: g = sum(ad_embed), softmax over T ----------
__global__ __launch_bounds__(256) void dien_attn_softmax(
    const int* __restrict__ ad_feature, const float* __restrict__ emb,
    const float* __restrict__ scores, float* __restrict__ attn) {
  __shared__ float sm[256];
  const int b = blockIdx.x, tid = threadIdx.x;
  float gp = (tid < En) ? emb[(size_t)ad_feature[b] * En + tid] : 0.f;
  sm[tid] = gp; __syncthreads();
  for (int s = 128; s > 0; s >>= 1) { if (tid < s) sm[tid] += sm[tid + s]; __syncthreads(); }
  const float g = sm[0]; __syncthreads();
  const float sv = (tid < Tn) ? scores[b * Tn + tid] * g : -3.4e38f;
  sm[tid] = sv; __syncthreads();
  for (int s = 128; s > 0; s >>= 1) { if (tid < s) sm[tid] = fmaxf(sm[tid], sm[tid + s]); __syncthreads(); }
  const float mx = sm[0]; __syncthreads();
  const float e = (tid < Tn) ? expf(sv - mx) : 0.f;
  sm[tid] = e; __syncthreads();
  for (int s = 128; s > 0; s >>= 1) { if (tid < s) sm[tid] += sm[tid + s]; __syncthreads(); }
  const float inv = 1.f / sm[0];
  if (tid < Tn) attn[b * Tn + tid] = e * inv;
}

// ---------- scan 2: AUGRU step ----------
__global__ __launch_bounds__(256) void dien_augru_step(
    const int* __restrict__ user_hist, const float* __restrict__ emb,
    const uint16_t* __restrict__ Wih, const uint16_t* __restrict__ Whh,
    const float* __restrict__ bih, const float* __restrict__ bhh,
    const float* __restrict__ wg, const float* __restrict__ bg,
    const float* __restrict__ attn,
    const float* __restrict__ h_in, float* __restrict__ h_out, int t) {
  __shared__ alignas(16) uint16_t xs[16 * En];
  __shared__ alignas(16) uint16_t hs[16 * En];
  __shared__ alignas(16) float gi_s[16 * Gn];
  __shared__ alignas(16) float gh_s[16 * Gn];
  __shared__ float red[256];
  __shared__ float rg_s[16];
  const int tid = threadIdx.x;
  const int b0  = blockIdx.x * 16;
  const int m   = tid >> 4;
  const int c   = (tid & 15) * 8;
  const int b   = b0 + m;

  __builtin_prefetch(Wih + tid * 192, 0, 3);
  __builtin_prefetch(Whh + tid * 192, 0, 3);

  {
    const int idx = user_hist[b * Tn + t];
    const float4* xr = reinterpret_cast<const float4*>(emb + (size_t)idx * En + c);
    float4 x0 = xr[0], x1 = xr[1];
    uint32_t* xd = reinterpret_cast<uint32_t*>(&xs[m * En + c]);
    xd[0] = pk2(x0.x, x0.y); xd[1] = pk2(x0.z, x0.w);
    xd[2] = pk2(x1.x, x1.y); xd[3] = pk2(x1.z, x1.w);
    const float4* hr = reinterpret_cast<const float4*>(h_in + b * Hn + c);
    float4 h0 = hr[0], h1 = hr[1];
    uint32_t* hd = reinterpret_cast<uint32_t*>(&hs[m * En + c]);
    hd[0] = pk2(h0.x, h0.y); hd[1] = pk2(h0.z, h0.w);
    hd[2] = pk2(h1.x, h1.y); hd[3] = pk2(h1.z, h1.w);
    // partial for rg = sigmoid(h . wg + bg)
    red[tid] = h0.x * wg[c + 0] + h0.y * wg[c + 1] + h0.z * wg[c + 2] + h0.w * wg[c + 3]
             + h1.x * wg[c + 4] + h1.y * wg[c + 5] + h1.z * wg[c + 6] + h1.w * wg[c + 7];
  }
  __syncthreads();
  gemm_tiles(xs, hs, Wih, Whh, bih, bhh, gi_s, gh_s);
  __syncthreads();
  if (tid < 16) {
    float s = 0.f;
#pragma unroll
    for (int i = 0; i < 16; ++i) s += red[tid * 16 + i];
    rg_s[tid] = sigmoidf_(s + bg[0]);
  }
  __syncthreads();

  const float a  = attn[b * Tn + t];
  const float rg = rg_s[m];
#pragma unroll
  for (int jj = 0; jj < 8; ++jj) {
    const int j = c + jj;
    const float r   = sigmoidf_(gi_s[m * Gn + j]       + gh_s[m * Gn + j]);
    const float z   = sigmoidf_(gi_s[m * Gn + 128 + j] + gh_s[m * Gn + 128 + j]);
    const float n   = tanhf(gi_s[m * Gn + 256 + j] + r * gh_s[m * Gn + 256 + j]);
    const float ho  = h_in[b * Hn + j];
    const float hgru = (1.f - z) * n + z * ho;
    const float h1v  = (1.f - rg) * ho + rg * hgru;
    h_out[b * Hn + j] = a * h1v + (1.f - a) * ho;
  }
}

// ---------- final MLP 128->64->32->1 ----------
__global__ __launch_bounds__(64) void dien_mlp(
    const float* __restrict__ h, const float* __restrict__ W1, const float* __restrict__ b1,
    const float* __restrict__ W2, const float* __restrict__ b2,
    const float* __restrict__ W3, const float* __restrict__ b3,
    float* __restrict__ out) {
  __shared__ float hr[128];
  __shared__ float a1[64];
  __shared__ float a2[32];
  const int b = blockIdx.x, tid = threadIdx.x;
  hr[tid]      = h[b * Hn + tid];
  hr[tid + 64] = h[b * Hn + tid + 64];
  __syncthreads();
  float acc = b1[tid];
  for (int k = 0; k < 128; ++k) acc += hr[k] * W1[tid * 128 + k];
  a1[tid] = fmaxf(acc, 0.f);
  __syncthreads();
  if (tid < 32) {
    float a = b2[tid];
    for (int k = 0; k < 64; ++k) a += a1[k] * W2[tid * 64 + k];
    a2[tid] = fmaxf(a, 0.f);
  }
  __syncthreads();
  if (tid == 0) {
    float a = b3[0];
    for (int k = 0; k < 32; ++k) a += a2[k] * W3[k];
    out[b] = a;
  }
}

// ---------- utility kernels ----------
__global__ void cvt_f32_bf16(const float* __restrict__ src, uint16_t* __restrict__ dst, int n) {
  const int i = blockIdx.x * blockDim.x + threadIdx.x;
  if (i < n) dst[i] = f2bf(src[i]);
}
__global__ void zero_f32(float* __restrict__ p, int n) {
  const int i = blockIdx.x * blockDim.x + threadIdx.x;
  if (i < n) p[i] = 0.f;
}

extern "C" void kernel_launch(void* const* d_in, const int* in_sizes, int n_in,
                              void* d_out, int out_size, void* d_ws, size_t ws_size,
                              hipStream_t stream) {
  (void)in_sizes; (void)n_in; (void)out_size; (void)ws_size;
  const int*   user_hist  = (const int*)  d_in[0];
  const int*   ad_feature = (const int*)  d_in[1];
  const float* emb   = (const float*)d_in[2];
  const float* Wih1  = (const float*)d_in[3];
  const float* Whh1  = (const float*)d_in[4];
  const float* bih1  = (const float*)d_in[5];
  const float* bhh1  = (const float*)d_in[6];
  const float* wa    = (const float*)d_in[7];
  const float* ba    = (const float*)d_in[8];
  const float* Wih2  = (const float*)d_in[9];
  const float* Whh2  = (const float*)d_in[10];
  const float* bih2  = (const float*)d_in[11];
  const float* bhh2  = (const float*)d_in[12];
  const float* wg    = (const float*)d_in[13];
  const float* bg    = (const float*)d_in[14];
  const float* W1    = (const float*)d_in[15];
  const float* b1    = (const float*)d_in[16];
  const float* W2    = (const float*)d_in[17];
  const float* b2    = (const float*)d_in[18];
  const float* W3    = (const float*)d_in[19];
  const float* b3    = (const float*)d_in[20];
  float* out = (float*)d_out;

  // workspace layout (bytes, 256-aligned)
  char* ws = (char*)d_ws;
  const int nW = Gn * 128;                       // 49152 elems, both Wih and Whh
  uint16_t* Wih1b = (uint16_t*)(ws + 0);
  uint16_t* Whh1b = (uint16_t*)(ws + 98304);
  uint16_t* Wih2b = (uint16_t*)(ws + 196608);
  uint16_t* Whh2b = (uint16_t*)(ws + 294912);
  float* hA     = (float*)(ws + 393216);         // Bsz*Hn f32
  float* hB     = (float*)(ws + 2490368);
  float* scores = (float*)(ws + 4587520);        // Bsz*Tn
  float* attn   = (float*)(ws + 7864320);        // Bsz*Tn

  cvt_f32_bf16<<<(nW + 255) / 256, 256, 0, stream>>>(Wih1, Wih1b, nW);
  cvt_f32_bf16<<<(nW + 255) / 256, 256, 0, stream>>>(Whh1, Whh1b, nW);
  cvt_f32_bf16<<<(nW + 255) / 256, 256, 0, stream>>>(Wih2, Wih2b, nW);
  cvt_f32_bf16<<<(nW + 255) / 256, 256, 0, stream>>>(Whh2, Whh2b, nW);
  zero_f32<<<(Bsz * Hn + 255) / 256, 256, 0, stream>>>(hA, Bsz * Hn);

  float* cur = hA; float* nxt = hB;
  for (int t = 0; t < Tn; ++t) {
    dien_gru1_step<<<Bsz / 16, 256, 0, stream>>>(
        user_hist, emb, Wih1b, Whh1b, bih1, bhh1, wa, ba, cur, nxt, scores, t);
    float* tmp = cur; cur = nxt; nxt = tmp;
  }
  // T even -> h_last lives in hA (== cur)

  dien_attn_softmax<<<Bsz, 256, 0, stream>>>(ad_feature, emb, scores, attn);

  for (int t = 0; t < Tn; ++t) {
    dien_augru_step<<<Bsz / 16, 256, 0, stream>>>(
        user_hist, emb, Wih2b, Whh2b, bih2, bhh2, wg, bg, attn, cur, nxt, t);
    float* tmp = cur; cur = nxt; nxt = tmp;
  }

  dien_mlp<<<Bsz, 64, 0, stream>>>(cur, W1, b1, W2, b2, W3, b3, out);
}
